// QuantumTransformerBlock_65481071399517
// MI455X (gfx1250) — matrix operand order repaired and checked by reading, not compiled, not run on hardware
//
#include <hip/hip_runtime.h>
#include <math.h>

// ---------------------------------------------------------------------------
// QuantumTransformerBlock for MI455X (gfx1250).
//   All GEMMs: bf16 WMMA (v_wmma_f32_16x16x32_bf16), NT layout (B = [N,K]),
//   async global->LDS tile streaming (global_load_async_to_lds_b128) with
//   ping-pong double buffering (software pipeline unrolled by 2, peeled tail),
//   fused bias/ReLU/bf16/transposed epilogues.
//   B=8, S=2048, E=1024, F=4096, NQ=8
// ---------------------------------------------------------------------------

typedef __attribute__((ext_vector_type(16))) __bf16          v16bf;
typedef __attribute__((ext_vector_type(8)))  float           v8f;
typedef __attribute__((ext_vector_type(8)))  unsigned short  v8u16;

__device__ __forceinline__ unsigned short f2bf(float f) {
  unsigned int u = __float_as_uint(f);
  unsigned int r = (u + 0x7FFFu + ((u >> 16) & 1u)) >> 16;  // RNE truncation
  return (unsigned short)r;
}

__device__ __forceinline__ v16bf ld_frag(const unsigned short* p0,
                                         const unsigned short* p1) {
  union { v8u16 u[2]; v16bf b; } cv;
  cv.u[0] = *(const v8u16*)p0;
  cv.u[1] = *(const v8u16*)p1;
  return cv.b;
}

// Async DMA of 16 bytes global -> LDS, tracked by ASYNCcnt (CDNA5 §10).
__device__ __forceinline__ void async_g2l_b128(unsigned lds_byte_off,
                                               const void* gptr) {
  asm volatile("global_load_async_to_lds_b128 %0, %1, off"
               :: "v"(lds_byte_off), "v"((unsigned long long)(size_t)gptr)
               : "memory");
}
__device__ __forceinline__ void wait_async_le4() {
  asm volatile("s_wait_asynccnt 0x4" ::: "memory");
}
__device__ __forceinline__ void wait_async_0() {
  asm volatile("s_wait_asynccnt 0x0" ::: "memory");
}

// ---------------------------------------------------------------------------
// NT bf16 WMMA GEMM: C[M,N] = alpha * A[M,K] @ B[N,K]^T + bias (+ReLU)
//   OUT_MODE: 0 = f32 [M,N], 1 = bf16 [M,N],
//             2 = bf16 batch-transposed: C[b][n][t] with b = m >> tshift,
//                 t = m & (2^tshift - 1), batch stride = strideC.
// Tile 128x128x32, 256 threads (8 waves in a 4x2 grid, each wave 32x64 ->
// 2 A-frags x 4 B-frags -> 8 WMMAs / K-step). Tiles streamed via async DMA
// into ping-pong LDS buffers; K-pipeline unrolled by 2 so buffer selection
// is static and the hot loop is branch-free.
// Requires: M,N multiples of 128; K multiple of 64 (nk even, >= 4).
// ---------------------------------------------------------------------------
template <int OUT_MODE, bool RELU>
__global__ __launch_bounds__(256)
void gemm_nt_bf16_wmma(const unsigned short* __restrict__ A,
                       const unsigned short* __restrict__ B,
                       const float* __restrict__ bias,
                       void* __restrict__ C,
                       int M, int N, int K,
                       long strideA, long strideB, long strideC,
                       float alpha, int tshift) {
  constexpr int BM = 128, BN = 128, BK = 32;
  constexpr int LDT = 40;  // padded K-stride in ushorts (80B: 16B-aligned, bank-spread)
  constexpr int BUF = BM * LDT;
  constexpr unsigned BUFB = BUF * 2;  // buffer stride in bytes
  __shared__ __align__(16) unsigned short As[2 * BUF];  // As[m][k]
  __shared__ __align__(16) unsigned short Bs[2 * BUF];  // Bs[n][k]

  const int bz = blockIdx.z;
  const unsigned short* Ab = A + (long)bz * strideA;
  const unsigned short* Bb = B + (long)bz * strideB;

  const int tile_m = blockIdx.y * BM;
  const int tile_n = blockIdx.x * BN;
  const int tid  = threadIdx.x;
  const int wave = tid >> 5;
  const int lane = tid & 31;
  const int wm = wave >> 1;       // 0..3 : 32-row stripe
  const int wn = wave & 1;        // 0..1 : 64-col half

  // Per-thread staging map: 2 chunks of 16B for each of A and B per K-step.
  const int row0 = tid >> 2;              // 0..63
  const int row1 = row0 + 64;             // 64..127
  const int c8   = (tid & 3) * 8;         // 0,8,16,24

  // Global source pointers (stepped by BK elements per stage).
  const unsigned short* a0 = Ab + (long)(tile_m + row0) * K + c8;
  const unsigned short* a1 = Ab + (long)(tile_m + row1) * K + c8;
  const unsigned short* b0 = Bb + (long)(tile_n + row0) * K + c8;
  const unsigned short* b1 = Bb + (long)(tile_n + row1) * K + c8;

  // LDS destination byte offsets for buffer 0 (buffer 1 = +BUFB).
  const unsigned ad0 = (unsigned)(size_t)(As + row0 * LDT + c8);
  const unsigned ad1 = (unsigned)(size_t)(As + row1 * LDT + c8);
  const unsigned bd0 = (unsigned)(size_t)(Bs + row0 * LDT + c8);
  const unsigned bd1 = (unsigned)(size_t)(Bs + row1 * LDT + c8);

  auto stage = [&](unsigned sel) {
    async_g2l_b128(ad0 + sel, a0);
    async_g2l_b128(ad1 + sel, a1);
    async_g2l_b128(bd0 + sel, b0);
    async_g2l_b128(bd1 + sel, b1);
    __builtin_prefetch(a0 + BK, 0, 0);  // speculative, dropped if invalid
    a0 += BK; a1 += BK; b0 += BK; b1 += BK;
  };

  v8f acc[2][4];
#pragma unroll
  for (int i = 0; i < 2; ++i)
#pragma unroll
    for (int j = 0; j < 4; ++j) acc[i][j] = (v8f)(0.0f);

  const int ahalf = (lane >> 4) * 8;
  const int bbase = (lane >> 4) * 8;
  const int lrow  = lane & 15;

  // One K-step of WMMAs out of LDS buffer (Ac, Bc).
  auto compute = [&](const unsigned short* Ac, const unsigned short* Bc) {
    // A 16x32 bf16: lanes 0-15 (row=lane) hold K{0..7,16..23};
    //               lanes 16-31 hold K{8..15,24..31}.
    v16bf afrag[2];
#pragma unroll
    for (int s = 0; s < 2; ++s) {
      const int mrow = wm * 32 + s * 16 + lrow;
      afrag[s] = ld_frag(Ac + mrow * LDT + ahalf,
                         Ac + mrow * LDT + 16 + ahalf);
    }
    // B 32x16 bf16: lanes 0-15 hold K{0..7,16..23} of col=lane; lanes 16-31 K{8..15,24..31}.
#pragma unroll
    for (int nt = 0; nt < 4; ++nt) {
      const int ncol = wn * 64 + nt * 16 + lrow;
      const v16bf bfrag = ld_frag(Bc + ncol * LDT + bbase,
                                  Bc + ncol * LDT + bbase + 16);
#pragma unroll
      for (int s = 0; s < 2; ++s) {
        acc[s][nt] = __builtin_amdgcn_wmma_f32_16x16x32_bf16(
            false, afrag[s], false, bfrag, (short)0, acc[s][nt], false, false);
      }
    }
  };

  const int nk = K / BK;  // even, >= 4

  // ---- software pipeline: stage tile0, then 2-unrolled steady state ----
  stage(0);
  for (int kt = 0; kt + 2 < nk; kt += 2) {
    stage(BUFB);           // tile kt+1 -> buf1, overlaps compute of buf0
    wait_async_le4();
    __syncthreads();
    compute(As, Bs);
    __syncthreads();
    stage(0);              // tile kt+2 -> buf0, overlaps compute of buf1
    wait_async_le4();
    __syncthreads();
    compute(As + BUF, Bs + BUF);
    __syncthreads();
  }
  // ---- peeled final pair ----
  stage(BUFB);
  wait_async_le4();
  __syncthreads();
  compute(As, Bs);
  wait_async_0();
  __syncthreads();
  compute(As + BUF, Bs + BUF);

  // ---- epilogue: D layout: VGPR r -> M=r (+8 for lanes 16-31), N=lane&15 ----
  const int rhi = (lane >> 4) ? 8 : 0;
#pragma unroll
  for (int s = 0; s < 2; ++s) {
    const int grow0 = tile_m + wm * 32 + s * 16 + rhi;
#pragma unroll
    for (int nt = 0; nt < 4; ++nt) {
      const int gc = tile_n + wn * 64 + nt * 16 + lrow;
      const float bv = bias ? bias[gc] : 0.0f;
#pragma unroll
      for (int r = 0; r < 8; ++r) {
        float v = acc[s][nt][r] * alpha + bv;
        if (RELU) v = fmaxf(v, 0.0f);
        const int grow = grow0 + r;
        if (OUT_MODE == 0) {
          ((float*)C)[(long)bz * strideC + (long)grow * N + gc] = v;
        } else if (OUT_MODE == 1) {
          ((unsigned short*)C)[(long)bz * strideC + (long)grow * N + gc] = f2bf(v);
        } else {  // bf16, batch-transposed: C[b][gc][t]
          const int b = grow >> tshift;
          const int t = grow & ((1 << tshift) - 1);
          ((unsigned short*)C)[(long)b * strideC + (long)gc * (1L << tshift) + t] = f2bf(v);
        }
      }
    }
  }
}

// ---------------------------------------------------------------------------
// Elementwise: f32 -> bf16 cast
// ---------------------------------------------------------------------------
__global__ __launch_bounds__(256)
void cast_f32_bf16(const float* __restrict__ in, unsigned short* __restrict__ out,
                   long n) {
  long i = (long)blockIdx.x * blockDim.x + threadIdx.x;
  if (i < n) out[i] = f2bf(in[i]);
}

// ---------------------------------------------------------------------------
// Transpose + cast: in f32 [R,C] -> out bf16 [C,R]. 32x32 LDS tiles.
// ---------------------------------------------------------------------------
__global__ __launch_bounds__(256)
void transpose_cast_f32_bf16(const float* __restrict__ in,
                             unsigned short* __restrict__ out, int R, int C) {
  __shared__ float tile[32][33];
  const int bx = blockIdx.x * 32;   // col base
  const int by = blockIdx.y * 32;   // row base
  const int tx = threadIdx.x & 31;
  const int ty = threadIdx.x >> 5;  // 0..7
#pragma unroll
  for (int j = 0; j < 4; ++j)
    tile[ty + j * 8][tx] = in[(long)(by + ty + j * 8) * C + bx + tx];
  __syncthreads();
#pragma unroll
  for (int j = 0; j < 4; ++j)
    out[(long)(bx + ty + j * 8) * R + by + tx] = f2bf(tile[tx][ty + j * 8]);
}

// ---------------------------------------------------------------------------
// out_bf16 = bf16( attn_out + cos(x + theta[e % 8]) ),  e = i % 1024
// ---------------------------------------------------------------------------
__global__ __launch_bounds__(256)
void qmeas_add_cast(const float* __restrict__ ao, const float* __restrict__ x,
                    const float* __restrict__ theta,
                    unsigned short* __restrict__ out, long n) {
  long i = (long)blockIdx.x * blockDim.x + threadIdx.x;
  if (i < n) {
    const int e = (int)(i & 1023);
    out[i] = f2bf(ao[i] + __cosf(x[i] + theta[e & 7]));
  }
}

// ---------------------------------------------------------------------------
// Row softmax over 2048 columns; writes bf16 probabilities.
// ---------------------------------------------------------------------------
__global__ __launch_bounds__(256)
void softmax_rows_bf16(const float* __restrict__ S, unsigned short* __restrict__ P) {
  constexpr int COLS = 2048, NPT = COLS / 256;
  __shared__ float red[256];
  const long row = blockIdx.x;
  const float* s = S + row * COLS;
  unsigned short* p = P + row * COLS;
  const int t = threadIdx.x;

  float v[NPT];
  float m = -INFINITY;
#pragma unroll
  for (int j = 0; j < NPT; ++j) {
    v[j] = s[t + j * 256];
    m = fmaxf(m, v[j]);
  }
  red[t] = m;
  __syncthreads();
  for (int off = 128; off > 0; off >>= 1) {
    if (t < off) red[t] = fmaxf(red[t], red[t + off]);
    __syncthreads();
  }
  m = red[0];
  __syncthreads();

  float sum = 0.0f;
#pragma unroll
  for (int j = 0; j < NPT; ++j) {
    v[j] = __expf(v[j] - m);
    sum += v[j];
  }
  red[t] = sum;
  __syncthreads();
  for (int off = 128; off > 0; off >>= 1) {
    if (t < off) red[t] += red[t + off];
    __syncthreads();
  }
  const float inv = __frcp_rn(red[0]);
#pragma unroll
  for (int j = 0; j < NPT; ++j) p[t + j * 256] = f2bf(v[j] * inv);
}

// ---------------------------------------------------------------------------
// LayerNorm(res + y) over E=1024; f32 out + optional bf16 out.
// ---------------------------------------------------------------------------
__global__ __launch_bounds__(256)
void ln_residual(const float* __restrict__ res, const float* __restrict__ y,
                 const float* __restrict__ g, const float* __restrict__ b,
                 float* __restrict__ out, unsigned short* __restrict__ out_b) {
  constexpr int E = 1024, NPT = E / 256;
  __shared__ float rs[256], rs2[256];
  const long row = blockIdx.x;
  const int t = threadIdx.x;
  const float* pr = res + row * E;
  const float* py = y + row * E;

  float x[NPT], s = 0.0f, s2 = 0.0f;
#pragma unroll
  for (int j = 0; j < NPT; ++j) {
    x[j] = pr[t + j * 256] + py[t + j * 256];
    s += x[j];
    s2 += x[j] * x[j];
  }
  rs[t] = s; rs2[t] = s2;
  __syncthreads();
  for (int off = 128; off > 0; off >>= 1) {
    if (t < off) { rs[t] += rs[t + off]; rs2[t] += rs2[t + off]; }
    __syncthreads();
  }
  const float mean = rs[0] * (1.0f / E);
  const float var  = rs2[0] * (1.0f / E) - mean * mean;
  const float rstd = rsqrtf(var + 1e-5f);
#pragma unroll
  for (int j = 0; j < NPT; ++j) {
    const int e = t + j * 256;
    const float v = (x[j] - mean) * rstd * g[e] + b[e];
    out[row * E + e] = v;
    if (out_b) out_b[row * E + e] = f2bf(v);
  }
}

// ---------------------------------------------------------------------------
// Host launcher
// ---------------------------------------------------------------------------
extern "C" void kernel_launch(void* const* d_in, const int* in_sizes, int n_in,
                              void* d_out, int out_size, void* d_ws, size_t ws_size,
                              hipStream_t stream) {
  (void)in_sizes; (void)n_in; (void)out_size; (void)ws_size;
  constexpr int  Bn = 8, S = 2048, E = 1024, F = 4096;
  constexpr long BS = (long)Bn * S;  // 16384 rows
  constexpr int  TSH = 11;           // log2(S)

  const float* x     = (const float*)d_in[0];
  const float* wq    = (const float*)d_in[1];
  const float* bq    = (const float*)d_in[2];
  const float* wk    = (const float*)d_in[3];
  const float* bk    = (const float*)d_in[4];
  const float* wv    = (const float*)d_in[5];
  const float* bv    = (const float*)d_in[6];
  const float* wc    = (const float*)d_in[7];
  const float* bc    = (const float*)d_in[8];
  const float* w1    = (const float*)d_in[9];
  const float* b1    = (const float*)d_in[10];
  const float* w2    = (const float*)d_in[11];
  const float* b2    = (const float*)d_in[12];
  const float* g1    = (const float*)d_in[13];
  const float* be1   = (const float*)d_in[14];
  const float* g2    = (const float*)d_in[15];
  const float* be2   = (const float*)d_in[16];
  const float* theta = (const float*)d_in[17];

  // ---- workspace layout ----
  size_t off = 0;
  char* base = (char*)d_ws;
  auto alloc = [&](size_t bytes) -> void* {
    void* p = base + off;
    off = (off + bytes + 255) & ~(size_t)255;
    return p;
  };
  unsigned short* xb   = (unsigned short*)alloc(BS * E * 2);       // x, bf16
  unsigned short* wqT  = (unsigned short*)alloc((long)E * E * 2);  // [out,in]
  unsigned short* wkT  = (unsigned short*)alloc((long)E * E * 2);
  unsigned short* wvT  = (unsigned short*)alloc((long)E * E * 2);
  unsigned short* wcT  = (unsigned short*)alloc((long)E * E * 2);
  unsigned short* w1T  = (unsigned short*)alloc((long)E * F * 2);  // [F,E]
  unsigned short* w2T  = (unsigned short*)alloc((long)F * E * 2);  // [E,F]
  unsigned short* qb   = (unsigned short*)alloc(BS * E * 2);       // [m,e]
  unsigned short* kb   = (unsigned short*)alloc(BS * E * 2);       // [m,e]
  unsigned short* vT   = (unsigned short*)alloc(BS * E * 2);       // [b][e][t]
  float*          sc   = (float*)alloc((long)Bn * S * S * 4);
  unsigned short* attnb= (unsigned short*)alloc((long)Bn * S * S * 2);
  float*          ao   = (float*)alloc(BS * E * 4);
  unsigned short* aob  = (unsigned short*)alloc(BS * E * 2);
  float*          proj = (float*)alloc(BS * E * 4);
  float*          h    = (float*)alloc(BS * E * 4);
  unsigned short* hb   = (unsigned short*)alloc(BS * E * 2);
  unsigned short* f1b  = (unsigned short*)alloc(BS * F * 2);
  float*          f2   = (float*)alloc(BS * E * 4);

  const dim3 blk256(256);
  auto ew = [](long n) { return dim3((unsigned)((n + 255) / 256)); };

  // ---- 1. cast x; transpose-cast all weights to [N,K] bf16 ----
  cast_f32_bf16<<<ew(BS * E), blk256, 0, stream>>>(x, xb, BS * E);
  transpose_cast_f32_bf16<<<dim3(E / 32, E / 32), blk256, 0, stream>>>(wq, wqT, E, E);
  transpose_cast_f32_bf16<<<dim3(E / 32, E / 32), blk256, 0, stream>>>(wk, wkT, E, E);
  transpose_cast_f32_bf16<<<dim3(E / 32, E / 32), blk256, 0, stream>>>(wv, wvT, E, E);
  transpose_cast_f32_bf16<<<dim3(E / 32, E / 32), blk256, 0, stream>>>(wc, wcT, E, E);
  transpose_cast_f32_bf16<<<dim3(F / 32, E / 32), blk256, 0, stream>>>(w1, w1T, E, F);
  transpose_cast_f32_bf16<<<dim3(E / 32, F / 32), blk256, 0, stream>>>(w2, w2T, F, E);

  // ---- 2. QKV projections (NT): q,k -> bf16 [m,e]; v -> bf16 transposed ----
  {
    dim3 grid(E / 128, BS / 128, 1);
    gemm_nt_bf16_wmma<1, false><<<grid, blk256, 0, stream>>>(
        xb, wqT, bq, qb, (int)BS, E, E, 0, 0, 0, 1.0f, TSH);
    gemm_nt_bf16_wmma<1, false><<<grid, blk256, 0, stream>>>(
        xb, wkT, bk, kb, (int)BS, E, E, 0, 0, 0, 1.0f, TSH);
    gemm_nt_bf16_wmma<2, false><<<grid, blk256, 0, stream>>>(
        xb, wvT, bv, vT, (int)BS, E, E, 0, 0, (long)E * S, 1.0f, TSH);
  }
  // ---- 3. scores = q @ k^T / sqrt(E): NT batched (k is [t,e] = [N,K]) ----
  {
    dim3 grid(S / 128, S / 128, Bn);
    gemm_nt_bf16_wmma<0, false><<<grid, blk256, 0, stream>>>(
        qb, kb, nullptr, sc, S, S, E,
        (long)S * E, (long)S * E, (long)S * S, 0.03125f, TSH);
  }
  // ---- 4. softmax rows -> bf16 attn ----
  softmax_rows_bf16<<<dim3((unsigned)(Bn * S)), blk256, 0, stream>>>(sc, attnb);
  // ---- 5. attn_out = attn @ v : NT with vT = [e][t] per batch ----
  {
    dim3 grid(E / 128, S / 128, Bn);
    gemm_nt_bf16_wmma<0, false><<<grid, blk256, 0, stream>>>(
        attnb, vT, nullptr, ao, S, E, S,
        (long)S * S, (long)E * S, (long)S * E, 1.0f, TSH);
  }
  // ---- 6. + quantum measurement cos(x + theta), cast to bf16 ----
  qmeas_add_cast<<<ew(BS * E), blk256, 0, stream>>>(ao, x, theta, aob, BS * E);
  // ---- 7. output projection + bias ----
  {
    dim3 grid(E / 128, BS / 128, 1);
    gemm_nt_bf16_wmma<0, false><<<grid, blk256, 0, stream>>>(
        aob, wcT, bc, proj, (int)BS, E, E, 0, 0, 0, 1.0f, TSH);
  }
  // ---- 8. LN1(x + proj) -> h (f32) + hb (bf16) ----
  ln_residual<<<dim3((unsigned)BS), blk256, 0, stream>>>(x, proj, g1, be1, h, hb);
  // ---- 9. FFN1: relu(h @ w1 + b1) -> bf16 ----
  {
    dim3 grid(F / 128, BS / 128, 1);
    gemm_nt_bf16_wmma<1, true><<<grid, blk256, 0, stream>>>(
        hb, w1T, b1, f1b, (int)BS, F, E, 0, 0, 0, 1.0f, TSH);
  }
  // ---- 10. FFN2: f1 @ w2 + b2 ----
  {
    dim3 grid(E / 128, BS / 128, 1);
    gemm_nt_bf16_wmma<0, false><<<grid, blk256, 0, stream>>>(
        f1b, w2T, b2, f2, (int)BS, E, F, 0, 0, 0, 1.0f, TSH);
  }
  // ---- 11. LN2(h + ffn) -> d_out ----
  ln_residual<<<dim3((unsigned)BS), blk256, 0, stream>>>(
      h, f2, g2, be2, (float*)d_out, nullptr);
}
